// MoEAttnIntersection3_13649406067385
// MI455X (gfx1250) — matrix operand order repaired and checked by reading, not compiled, not run on hardware
//
#include <hip/hip_runtime.h>
#include <hip/hip_bf16.h>
#include <stdint.h>

typedef __attribute__((ext_vector_type(16))) _Float16 v16h;
typedef __attribute__((ext_vector_type(8)))  _Float16 v8h;
typedef __attribute__((ext_vector_type(4)))  _Float16 v4h;
typedef __attribute__((ext_vector_type(8)))  float    v8f;

#define ACT_NONE  0
#define ACT_RELU  1
#define ACT_LRELU 2

// ---------------------------------------------------------------------------
// CDNA5 async global->LDS copy (ASYNCcnt-tracked), 16 bytes per lane.
// lds_off = byte offset into LDS (low 32 bits of generic shared address).
// ---------------------------------------------------------------------------
__device__ __forceinline__ void async_ld_b128(uint32_t lds_off, const _Float16* gaddr) {
    asm volatile("global_load_async_to_lds_b128 %0, %1, off"
                 :: "v"(lds_off), "v"(gaddr)
                 : "memory");
}
__device__ __forceinline__ void wait_async0() {
    asm volatile("s_wait_asynccnt 0" ::: "memory");
}
__device__ __forceinline__ uint32_t lds_off_of(const void* p) {
    return (uint32_t)(uintptr_t)p;   // generic shared addr: low 32 bits = LDS offset
}

// ---------------------------------------------------------------------------
// WMMA GEMM, all-f16 operands, f32 accumulate:
//   C[:, colOff+n] = epi( A(MxK,f16) @ W(NxK,f16)^T + bias )
// Block: 256 threads = 8 waves; tile 128(M) x 64(N); k-step 32.
// Wave (wave&3 -> 32 rows, wave>>2 -> 32 cols): 4 WMMA accumulators.
// Staging: double-buffered LDS filled by global_load_async_to_lds_b128,
// pipelined one k-step ahead (ASYNCcnt).
// M must be a multiple of 128, K a multiple of 32. N arbitrary (row-clamped).
// ---------------------------------------------------------------------------
__global__ void __launch_bounds__(256)
gemm_wmma(const _Float16* __restrict__ A,    // M x K row-major
          const _Float16* __restrict__ W,    // N x K row-major
          const float* __restrict__ bias,    // N (f32)
          const float* __restrict__ resid,   // f32, indexed like C, or null
          const float* __restrict__ rowScale,// per-row scale (stride scaleStride) or null
          int scaleStride,
          float*    __restrict__ C,          // f32 output or null
          _Float16* __restrict__ C16,        // f16 output or null
          int ldc, int colOff,
          int M, int N, int K, int act)
{
    __shared__ __align__(16) _Float16 lsA[2][128][40];  // 32-wide k tile, pad to 40
    __shared__ __align__(16) _Float16 lsW[2][64][40];

    const int m0   = blockIdx.y * 128;
    const int n0   = blockIdx.x * 64;
    const int tid  = threadIdx.x;
    const int lane = tid & 31;
    const int wave = tid >> 5;
    const int wm   = (wave & 3) * 32;
    const int wn   = (wave >> 2) * 32;
    const int l15  = lane & 15;
    const int hi   = lane >> 4;
    // A 16x32 f16 fragment: lanes 0-15 = K0..15, lanes 16-31 = K16..31 (contiguous)
    const int kA0  = hi ? 16 : 0;
    const int kA1  = hi ? 24 : 8;
    // B 32x16 f16 fragment: lanes 0-15 = K0..15, lanes 16-31 = K16..31 (contiguous)
    const int kB   = hi ? 16 : 0;

    // staging assignment: A tile 128x32 halves -> 2x b128/thread; W tile 64x32 -> 1x b128/thread
    const int aRow = tid >> 1;
    const int aCol = (tid & 1) * 16;
    const int wRow = tid >> 2;
    const int wCol = (tid & 3) * 8;
    const int wRowC = (n0 + wRow < N) ? (n0 + wRow) : (N - 1);  // clamp: no OOB, cols>=N discarded

    const _Float16* Ag = A + (size_t)(m0 + aRow) * K + aCol;
    const _Float16* Wg = W + (size_t)wRowC * K + wCol;

    uint32_t lA0 = lds_off_of(&lsA[0][aRow][aCol]);
    uint32_t lA1 = lds_off_of(&lsA[1][aRow][aCol]);
    uint32_t lW0 = lds_off_of(&lsW[0][wRow][wCol]);
    uint32_t lW1 = lds_off_of(&lsW[1][wRow][wCol]);

    v8f acc00 = {}, acc01 = {}, acc10 = {}, acc11 = {};

    const int nk = K >> 5;

    // prefetch k-tile 0 into buffer 0
    async_ld_b128(lA0,      Ag);
    async_ld_b128(lA0 + 16, Ag + 8);
    async_ld_b128(lW0,      Wg);

    for (int t = 0; t < nk; ++t) {
        const int buf = t & 1;
        wait_async0();       // own async copies for tile t landed
        __syncthreads();     // everyone's landed; everyone done reading other buffer

        if (t + 1 < nk) {    // issue tile t+1 into the other buffer (overlaps WMMAs below)
            const _Float16* Agn = Ag + (size_t)(t + 1) * 32;
            const _Float16* Wgn = Wg + (size_t)(t + 1) * 32;
            uint32_t dA = buf ? lA0 : lA1;
            uint32_t dW = buf ? lW0 : lW1;
            async_ld_b128(dA,      Agn);
            async_ld_b128(dA + 16, Agn + 8);
            async_ld_b128(dW,      Wgn);
        }

        v16h a0, a1, b0, b1;
        ((v8h*)&a0)[0] = *(const v8h*)&lsA[buf][wm + l15][kA0];
        ((v8h*)&a0)[1] = *(const v8h*)&lsA[buf][wm + l15][kA1];
        ((v8h*)&a1)[0] = *(const v8h*)&lsA[buf][wm + 16 + l15][kA0];
        ((v8h*)&a1)[1] = *(const v8h*)&lsA[buf][wm + 16 + l15][kA1];
        b0 = *(const v16h*)&lsW[buf][wn + l15][kB];
        b1 = *(const v16h*)&lsW[buf][wn + 16 + l15][kB];

        acc00 = __builtin_amdgcn_wmma_f32_16x16x32_f16(false, a0, false, b0, (short)0, acc00, false, false);
        acc01 = __builtin_amdgcn_wmma_f32_16x16x32_f16(false, a0, false, b1, (short)0, acc01, false, false);
        acc10 = __builtin_amdgcn_wmma_f32_16x16x32_f16(false, a1, false, b0, (short)0, acc10, false, false);
        acc11 = __builtin_amdgcn_wmma_f32_16x16x32_f16(false, a1, false, b1, (short)0, acc11, false, false);
    }

    // Epilogue. C/D layout: VGPR j -> row = j + (lane>=16 ? 8 : 0), col = lane%16.
    #pragma unroll
    for (int mi = 0; mi < 2; ++mi) {
        #pragma unroll
        for (int ni = 0; ni < 2; ++ni) {
            v8f acc = mi ? (ni ? acc11 : acc10) : (ni ? acc01 : acc00);
            int col = n0 + wn + ni * 16 + l15;
            if (col < N) {
                float bv = bias ? bias[col] : 0.f;
                #pragma unroll
                for (int j = 0; j < 8; ++j) {
                    int row = m0 + wm + mi * 16 + hi * 8 + j;
                    float v = acc[j] + bv;
                    if (act == ACT_RELU)       v = fmaxf(v, 0.f);
                    else if (act == ACT_LRELU) v = v >= 0.f ? v : 0.01f * v;
                    if (rowScale) v *= rowScale[(size_t)row * scaleStride];
                    size_t off = (size_t)row * ldc + colOff + col;
                    if (resid) v += resid[off];
                    if (C)   C[off]   = v;
                    if (C16) C16[off] = (_Float16)v;
                }
            }
        }
    }
}

// ---------------------------------------------------------------------------
// f32 -> f16 conversion (weights pre-pass / activation snapshots). n % 4 == 0.
// ---------------------------------------------------------------------------
__global__ void __launch_bounds__(256)
cvt_f16_kernel(const float* __restrict__ x, _Float16* __restrict__ y, long n)
{
    long i = ((long)blockIdx.x * 256 + threadIdx.x) * 4;
    if (i >= n) return;
    float4 v = *(const float4*)(x + i);
    v4h h;
    h[0] = (_Float16)v.x; h[1] = (_Float16)v.y;
    h[2] = (_Float16)v.z; h[3] = (_Float16)v.w;
    *(v4h*)(y + i) = h;
}

// ---------------------------------------------------------------------------
// LayerNorm over D=512, one wave32 per row, f32 in -> f16 out.
// ---------------------------------------------------------------------------
__global__ void __launch_bounds__(256)
ln_kernel(const float* __restrict__ x, const float* __restrict__ s,
          const float* __restrict__ b, _Float16* __restrict__ y, int rows)
{
    int wave = threadIdx.x >> 5, lane = threadIdx.x & 31;
    int row = blockIdx.x * 8 + wave;
    if (row >= rows) return;
    const float* xr = x + (size_t)row * 512;
    float vals[16];
    float sum = 0.f, sq = 0.f;
    #pragma unroll
    for (int i = 0; i < 16; ++i) {
        float v = xr[lane + i * 32];
        vals[i] = v; sum += v; sq += v * v;
    }
    #pragma unroll
    for (int off = 16; off > 0; off >>= 1) {
        sum += __shfl_xor(sum, off, 32);
        sq  += __shfl_xor(sq,  off, 32);
    }
    float m   = sum * (1.f / 512.f);
    float var = sq * (1.f / 512.f) - m * m;
    float inv = rsqrtf(var + 1e-5f);
    _Float16* yr = y + (size_t)row * 512;
    #pragma unroll
    for (int i = 0; i < 16; ++i) {
        int c = lane + i * 32;
        yr[c] = (_Float16)((vals[i] - m) * inv * s[c] + b[c]);
    }
}

// ---------------------------------------------------------------------------
// Split interleaved (2B x 512) inproj output into tgt(f32) / mem(f16) + pos.
// ---------------------------------------------------------------------------
__global__ void split_pos_kernel(const float* __restrict__ xbuf,
                                 const float* __restrict__ pos,
                                 float* __restrict__ tgt, _Float16* __restrict__ mem16,
                                 int n)
{
    int idx = blockIdx.x * blockDim.x + threadIdx.x;
    if (idx >= n) return;
    int b = idx >> 9, c = idx & 511;
    tgt[idx]   = xbuf[(size_t)(2 * b)     * 512 + c] + pos[c];
    mem16[idx] = (_Float16)(xbuf[(size_t)(2 * b + 1) * 512 + c] + pos[512 + c]);
}

// ---------------------------------------------------------------------------
// Top-2-of-8 softmax gating (jax.lax.top_k tie-breaks on lowest index).
// ---------------------------------------------------------------------------
__global__ void gate_kernel(const float* __restrict__ logits,
                            float* __restrict__ gates, int rows)
{
    int r = blockIdx.x * blockDim.x + threadIdx.x;
    if (r >= rows) return;
    float l[8];
    #pragma unroll
    for (int i = 0; i < 8; ++i) l[i] = logits[(size_t)r * 8 + i];
    int i0 = 0;
    #pragma unroll
    for (int i = 1; i < 8; ++i) if (l[i] > l[i0]) i0 = i;
    int i1 = (i0 == 0) ? 1 : 0;
    #pragma unroll
    for (int i = 0; i < 8; ++i) if (i != i0 && l[i] > l[i1]) i1 = i;
    float e1 = __expf(l[i1] - l[i0]);
    float p0 = 1.f / (1.f + e1);
    float p1 = e1 * p0;
    #pragma unroll
    for (int i = 0; i < 8; ++i) gates[(size_t)r * 8 + i] = 0.f;
    gates[(size_t)r * 8 + i0] = p0;
    gates[(size_t)r * 8 + i1] = p1;
}

// ---------------------------------------------------------------------------
// Host orchestration.  Seq len == 1 => attention softmax over one key == 1.0,
// so MHA(q_in, kv) == (kv@Wv^T+bv)@Wout^T+bout; Q/K projections and ln2 dead.
// ---------------------------------------------------------------------------
extern "C" void kernel_launch(void* const* d_in, const int* in_sizes, int n_in,
                              void* d_out, int out_size, void* d_ws, size_t ws_size,
                              hipStream_t stream)
{
    constexpr int B = 8192, DL = 512, DFF = 2048, E = 8, HID = 1024;
    constexpr int CATD = DL + E * DL;   // 4608

    const float* src     = (const float*)d_in[0];
    const float* piw     = (const float*)d_in[1];
    const float* pib     = (const float*)d_in[2];
    const float* pos     = (const float*)d_in[3];
    const float* ln1_s   = (const float*)d_in[4];
    const float* ln1_b   = (const float*)d_in[5];
    // d_in[6], d_in[7]: ln2_* -> dead (feeds only the dead Q path)
    const float* ln3_s   = (const float*)d_in[8];
    const float* ln3_b   = (const float*)d_in[9];
    const float* sa_in_w = (const float*)d_in[10];
    const float* sa_in_b = (const float*)d_in[11];
    const float* sa_out_w= (const float*)d_in[12];
    const float* sa_out_b= (const float*)d_in[13];
    const float* ca_in_w = (const float*)d_in[14];
    const float* ca_in_b = (const float*)d_in[15];
    const float* ca_out_w= (const float*)d_in[16];
    const float* ca_out_b= (const float*)d_in[17];
    const float* ff1_w   = (const float*)d_in[18];
    const float* ff1_b   = (const float*)d_in[19];
    const float* ff2_w   = (const float*)d_in[20];
    const float* ff2_b   = (const float*)d_in[21];
    const float* r1_w    = (const float*)d_in[22];
    const float* r1_b    = (const float*)d_in[23];
    const float* r2_w    = (const float*)d_in[24];
    const float* r2_b    = (const float*)d_in[25];
    const float* se1_w   = (const float*)d_in[26];
    const float* se1_b   = (const float*)d_in[27];
    const float* se2_w   = (const float*)d_in[28];
    const float* se2_b   = (const float*)d_in[29];
    const float* se3_w   = (const float*)d_in[30];
    const float* se3_b   = (const float*)d_in[31];
    const float* te1_w   = (const float*)d_in[32];
    const float* te1_b   = (const float*)d_in[33];
    const float* te2_w   = (const float*)d_in[34];
    const float* te2_b   = (const float*)d_in[35];
    const float* te3_w   = (const float*)d_in[36];
    const float* te3_b   = (const float*)d_in[37];
    const float* po_w    = (const float*)d_in[38];
    const float* po_b    = (const float*)d_in[39];

    // ---- workspace layout (256B-aligned sub-allocations) ----
    char* p = (char*)d_ws;
    auto allocF = [&](size_t n) {
        float* r = (float*)p; p += ((n * 4 + 255) & ~(size_t)255); return r;
    };
    auto allocH = [&](size_t n) {
        _Float16* r = (_Float16*)p; p += ((n * 2 + 255) & ~(size_t)255); return r;
    };

    float*    tgt     = allocF((size_t)B * DL);
    float*    xbuf    = allocF((size_t)2 * B * DL);
    float*    logitsW = allocF((size_t)B * E);
    float*    gatesW  = allocF((size_t)B * E);
    _Float16* src16   = allocH((size_t)2 * B * DL);
    _Float16* tgt16   = allocH((size_t)B * DL);
    _Float16* mem16   = allocH((size_t)B * DL);
    _Float16* h16     = allocH((size_t)B * DL);
    _Float16* a1h     = allocH((size_t)B * DFF);
    _Float16* a2h     = allocH((size_t)B * DL);
    _Float16* cat16   = allocH((size_t)B * CATD);
    // f16 weights
    _Float16* w_piw   = allocH((size_t)DL * DL);
    _Float16* w_sa_in = allocH((size_t)6 * 3 * DL * DL);
    _Float16* w_sa_out= allocH((size_t)6 * DL * DL);
    _Float16* w_ca_in = allocH((size_t)6 * 3 * DL * DL);
    _Float16* w_ca_out= allocH((size_t)6 * DL * DL);
    _Float16* w_ff1   = allocH((size_t)6 * DFF * DL);
    _Float16* w_ff2   = allocH((size_t)6 * DL * DFF);
    _Float16* w_r1    = allocH((size_t)256 * DL);
    _Float16* w_r2    = allocH((size_t)E * 256);
    _Float16* w_se1   = allocH((size_t)HID * DL);
    _Float16* w_se2   = allocH((size_t)DL * HID);
    _Float16* w_se3   = allocH((size_t)DL * DL);
    _Float16* w_te1   = allocH((size_t)E * HID * DL);
    _Float16* w_te2   = allocH((size_t)E * DL * HID);
    _Float16* w_te3   = allocH((size_t)E * DL * DL);
    _Float16* w_po    = allocH((size_t)DL * CATD);

    auto cvt = [&](const float* x, _Float16* y, long n) {
        cvt_f16_kernel<<<(unsigned)((n / 4 + 255) / 256), 256, 0, stream>>>(x, y, n);
    };
    auto gemm = [&](const _Float16* A, const _Float16* W, const float* bias,
                    const float* resid, const float* rowScale, int ss,
                    float* C, _Float16* C16, int ldc, int colOff,
                    int M, int N, int K, int act) {
        dim3 grid((N + 63) / 64, M / 128);
        gemm_wmma<<<grid, 256, 0, stream>>>(A, W, bias, resid, rowScale, ss,
                                            C, C16, ldc, colOff, M, N, K, act);
    };

    // ---- pre-pass: f16 weights + f16 src ----
    cvt(piw,      w_piw,    (long)DL * DL);
    cvt(sa_in_w,  w_sa_in,  (long)6 * 3 * DL * DL);
    cvt(sa_out_w, w_sa_out, (long)6 * DL * DL);
    cvt(ca_in_w,  w_ca_in,  (long)6 * 3 * DL * DL);
    cvt(ca_out_w, w_ca_out, (long)6 * DL * DL);
    cvt(ff1_w,    w_ff1,    (long)6 * DFF * DL);
    cvt(ff2_w,    w_ff2,    (long)6 * DL * DFF);
    cvt(r1_w,     w_r1,     (long)256 * DL);
    cvt(r2_w,     w_r2,     (long)E * 256);
    cvt(se1_w,    w_se1,    (long)HID * DL);
    cvt(se2_w,    w_se2,    (long)DL * HID);
    cvt(se3_w,    w_se3,    (long)DL * DL);
    cvt(te1_w,    w_te1,    (long)E * HID * DL);
    cvt(te2_w,    w_te2,    (long)E * DL * HID);
    cvt(te3_w,    w_te3,    (long)E * DL * DL);
    cvt(po_w,     w_po,     (long)DL * CATD);
    cvt(src,      src16,    (long)2 * B * DL);

    // ---- input projection + positional embedding ----
    gemm(src16, w_piw, pib, nullptr, nullptr, 0, xbuf, nullptr, DL, 0, 2 * B, DL, DL, ACT_NONE);
    split_pos_kernel<<<(B * DL + 255) / 256, 256, 0, stream>>>(xbuf, pos, tgt, mem16, B * DL);

    // ---- 6 decoder layers (attention collapsed to V/out projections) ----
    for (int i = 0; i < 6; ++i) {
        const _Float16* sa_wv = w_sa_in + (size_t)i * 3 * DL * DL + (size_t)2 * DL * DL;
        const float*    sa_bv = sa_in_b + (size_t)i * 3 * DL + 2 * DL;
        const _Float16* ca_wv = w_ca_in + (size_t)i * 3 * DL * DL + (size_t)2 * DL * DL;
        const float*    ca_bv = ca_in_b + (size_t)i * 3 * DL + 2 * DL;

        // self-attn: tgt += ((LN1(tgt) @ Wv^T + bv) @ Wout^T + bout)
        ln_kernel<<<B / 8, 256, 0, stream>>>(tgt, ln1_s + (size_t)i * DL, ln1_b + (size_t)i * DL, h16, B);
        gemm(h16, sa_wv, sa_bv, nullptr, nullptr, 0, nullptr, a2h, DL, 0, B, DL, DL, ACT_NONE);
        gemm(a2h, w_sa_out + (size_t)i * DL * DL, sa_out_b + (size_t)i * DL,
             tgt, nullptr, 0, tgt, nullptr, DL, 0, B, DL, DL, ACT_NONE);

        // cross-attn: tgt += ((mem @ Wv^T + bv) @ Wout^T + bout)   (ln2 dead)
        gemm(mem16, ca_wv, ca_bv, nullptr, nullptr, 0, nullptr, a2h, DL, 0, B, DL, DL, ACT_NONE);
        gemm(a2h, w_ca_out + (size_t)i * DL * DL, ca_out_b + (size_t)i * DL,
             tgt, nullptr, 0, tgt, nullptr, DL, 0, B, DL, DL, ACT_NONE);

        // FFN: tgt += relu(LN3(tgt) @ W1^T + b1) @ W2^T + b2
        ln_kernel<<<B / 8, 256, 0, stream>>>(tgt, ln3_s + (size_t)i * DL, ln3_b + (size_t)i * DL, h16, B);
        gemm(h16, w_ff1 + (size_t)i * DFF * DL, ff1_b + (size_t)i * DFF,
             nullptr, nullptr, 0, nullptr, a1h, DFF, 0, B, DFF, DL, ACT_RELU);
        gemm(a1h, w_ff2 + (size_t)i * DL * DFF, ff2_b + (size_t)i * DL,
             tgt, nullptr, 0, tgt, nullptr, DL, 0, B, DL, DFF, ACT_NONE);
    }

    // ---- f16 snapshot of final tgt for the heads ----
    cvt(tgt, tgt16, (long)B * DL);

    // ---- shared head -> cat[:, 0:512] ----
    gemm(tgt16, w_se1, se1_b, nullptr, nullptr, 0, nullptr, a1h, HID, 0, B, HID, DL, ACT_LRELU);
    gemm(a1h,   w_se2, se2_b, nullptr, nullptr, 0, nullptr, a2h, DL, 0, B, DL, HID, ACT_LRELU);
    gemm(a2h,   w_se3, se3_b, nullptr, nullptr, 0, nullptr, cat16, CATD, 0, B, DL, DL, ACT_NONE);

    // ---- router + top-2 softmax gating ----
    gemm(tgt16, w_r1, r1_b, nullptr, nullptr, 0, nullptr, a2h, 256, 0, B, 256, DL, ACT_LRELU);
    gemm(a2h,   w_r2, r2_b, nullptr, nullptr, 0, logitsW, nullptr, E, 0, B, E, 256, ACT_NONE);
    gate_kernel<<<(B + 255) / 256, 256, 0, stream>>>(logitsW, gatesW, B);

    // ---- experts, gate-scaled into cat[:, 512 + e*512 : ...] ----
    for (int e = 0; e < E; ++e) {
        gemm(tgt16, w_te1 + (size_t)e * HID * DL, te1_b + (size_t)e * HID,
             nullptr, nullptr, 0, nullptr, a1h, HID, 0, B, HID, DL, ACT_LRELU);
        gemm(a1h, w_te2 + (size_t)e * DL * HID, te2_b + (size_t)e * DL,
             nullptr, nullptr, 0, nullptr, a2h, DL, 0, B, DL, HID, ACT_LRELU);
        gemm(a2h, w_te3 + (size_t)e * DL * DL, te3_b + (size_t)e * DL,
             nullptr, gatesW + e, E, nullptr, cat16, CATD, DL + e * DL, B, DL, DL, ACT_NONE);
    }

    // ---- final projection ----
    gemm(cat16, w_po, po_b, nullptr, nullptr, 0, (float*)d_out, nullptr, DL, 0, B, DL, CATD, ACT_NONE);
}
